// LossFunction_48704929137325
// MI455X (gfx1250) — matrix operand order repaired and checked by reading, not compiled, not run on hardware
//
#include <hip/hip_runtime.h>
#include <hip/hip_bf16.h>
#include <math.h>

#define N_SPK 2048
#define G_UTT 10
#define D_FEAT 512
#define NG (N_SPK * G_UTT)   // 20480 rows
#define EPSN 1e-8f           // normalization eps (torch cosine_similarity default)

typedef __attribute__((ext_vector_type(16))) _Float16 v16h;
typedef __attribute__((ext_vector_type(4)))  _Float16 v4h;
typedef __attribute__((ext_vector_type(2)))  _Float16 v2h;
typedef __attribute__((ext_vector_type(8)))  float    v8f;

// ---------------------------------------------------------------------------
// Kernel 0: zero the (loss, correct) accumulators every launch (deterministic)
// ---------------------------------------------------------------------------
__global__ void ge2e_init_kernel(float* accum) {
    if (threadIdx.x < 2) accum[threadIdx.x] = 0.0f;
}

// ---------------------------------------------------------------------------
// Kernel 1: per-speaker prep. One block (256 thr) per speaker n.
//   inv_x[n*g+k] = 1/max(||x[n,k]||, eps)
//   diag[n*g+k]  = cos(xn[n,k], en[n,k])   (leave-one-out cosine)
//   Bmat[n][d]   = f16 normalized centroid (WMMA B operand)
// ---------------------------------------------------------------------------
__global__ __launch_bounds__(256)
void ge2e_prep_kernel(const float* __restrict__ x,
                      float* __restrict__ inv_x,
                      float* __restrict__ diag,
                      _Float16* __restrict__ Bmat) {
    const int n   = blockIdx.x;
    const int tid = threadIdx.x;
    const int f0  = tid * 2;
    const float* xs = x + (size_t)n * G_UTT * D_FEAT;

    float xg0[G_UTT], xg1[G_UTT];
    float sum0 = 0.f, sum1 = 0.f;
#pragma unroll
    for (int g = 0; g < G_UTT; ++g) {
        float2 v = *(const float2*)(xs + (size_t)g * D_FEAT + f0);
        xg0[g] = v.x; xg1[g] = v.y;
        sum0 += v.x;  sum1 += v.y;
    }
    const float cent0 = sum0 * (1.0f / G_UTT);
    const float cent1 = sum1 * (1.0f / G_UTT);

    // partials: [0..9]=||x_g||^2  [10..19]=||exc_g||^2  [20..29]=x_g.exc_g  [30]=||cent||^2
    float part[31];
#pragma unroll
    for (int g = 0; g < G_UTT; ++g) {
        const float e0 = (sum0 - xg0[g]) * (1.0f / (G_UTT - 1));
        const float e1 = (sum1 - xg1[g]) * (1.0f / (G_UTT - 1));
        part[g]      = xg0[g] * xg0[g] + xg1[g] * xg1[g];
        part[10 + g] = e0 * e0 + e1 * e1;
        part[20 + g] = xg0[g] * e0 + xg1[g] * e1;
    }
    part[30] = cent0 * cent0 + cent1 * cent1;

#pragma unroll
    for (int m = 16; m >= 1; m >>= 1) {
#pragma unroll
        for (int i = 0; i < 31; ++i) part[i] += __shfl_xor(part[i], m);
    }

    __shared__ float red[8][31];
    __shared__ float fin[31];
    const int wave = tid >> 5, lane = tid & 31;
    if (lane == 0) {
        for (int i = 0; i < 31; ++i) red[wave][i] = part[i];
    }
    __syncthreads();
    if (tid < 31) {
        float s = 0.f;
        for (int wv = 0; wv < 8; ++wv) s += red[wv][tid];
        fin[tid] = s;
    }
    __syncthreads();

    const float invc = 1.0f / fmaxf(sqrtf(fin[30]), EPSN);
    v2h bv;
    bv.x = (_Float16)(cent0 * invc);
    bv.y = (_Float16)(cent1 * invc);
    *(v2h*)(&Bmat[(size_t)n * D_FEAT + f0]) = bv;

    if (tid < G_UTT) {
        const float ivx = 1.0f / fmaxf(sqrtf(fin[tid]), EPSN);
        const float ive = 1.0f / fmaxf(sqrtf(fin[10 + tid]), EPSN);
        inv_x[n * G_UTT + tid] = ivx;
        diag[n * G_UTT + tid]  = fin[20 + tid] * ivx * ive;
    }
}

// ---------------------------------------------------------------------------
// Kernel 2: fused WMMA GEMM (xn . cn^T) + diag substitution + clip/scale +
// fixed-shift softmax sum + argmax + loss/precision accumulation.
//
// Block: 256 threads = 8 waves = 2 row-tiles x 4 column-groups.
// A fragments (16 x v16h = full K depth) live in registers; the inner loop
// issues 2 ds_load_b128 + 1 v_wmma per K-step, with the B fragment
// double-buffered in registers so DS latency pipelines across K-steps.
// ---------------------------------------------------------------------------
#define ROWS_PER_BLK 32
#define CGROUPS 4
#define TILES_PER_CG 32   // 2048/16/4
#define KSTEPS 16         // 512 / 32

__global__ __launch_bounds__(256)
void ge2e_gemm_loss_kernel(const float* __restrict__ x,
                           const float* __restrict__ inv_x,
                           const float* __restrict__ diag,
                           const _Float16* __restrict__ Bmat,
                           const float* __restrict__ wp,
                           const float* __restrict__ bp,
                           float* __restrict__ accum) {
    // 64 KB shared pool: first 32 KB doubles as the A staging area (32x512 f16),
    // then reused as 4 x (16x512 f16) B tiles, one per column group.
    __shared__ uint4 smem4[(CGROUPS * 16 * D_FEAT * 2) / 16];
    __shared__ float smax[ROWS_PER_BLK][CGROUPS];
    __shared__ float ssum[ROWS_PER_BLK][CGROUPS];
    __shared__ int   sarg[ROWS_PER_BLK][CGROUPS];
    __shared__ float lred[ROWS_PER_BLK];
    __shared__ float cred[ROWS_PER_BLK];

    _Float16* ldsA = (_Float16*)smem4;

    const int tid  = threadIdx.x;
    const int wv   = tid >> 5;
    const int lane = tid & 31;
    const int cg   = wv >> 1;      // 0..3 column group
    const int rt   = wv & 1;       // 0..1 row tile
    const int hi   = lane >> 4;    // lane half (K / M split per ISA layouts)
    const int ln   = lane & 15;

    const int row0 = blockIdx.x * ROWS_PER_BLK;
    const float w  = wp[0];
    const float bb = bp[0];
    const float shift = w + bb;    // upper bound of logits (cos <= 1)

    // ---- stage A: 32 rows of x, scaled by inv_norm, converted to f16 ----
    for (int i = 0; i < 16; ++i) {
        const int q = tid + i * 256;
        const int e = q * 4;
        const int r = e >> 9;        // /512
        const int c = e & 511;
        float4 v = *(const float4*)(x + (size_t)(row0 + r) * D_FEAT + c);
        const float s = inv_x[row0 + r];
        v4h hv;
        hv.x = (_Float16)(v.x * s); hv.y = (_Float16)(v.y * s);
        hv.z = (_Float16)(v.z * s); hv.w = (_Float16)(v.w * s);
        *(v4h*)(&ldsA[(size_t)r * D_FEAT + c]) = hv;
    }
    __syncthreads();

    // ---- hoist this wave's A fragments into registers (full K depth) ----
    // A frag layout (ISA 7.12.2): lanes<16 hold K {0..7,16..23}; lanes>=16 {8..15,24..31}
    const _Float16* arow = ldsA + (size_t)(rt * 16 + ln) * D_FEAT;
    v16h afr[KSTEPS];
#pragma unroll
    for (int kb = 0; kb < KSTEPS; ++kb) {
        union { uint4 q[2]; v16h v; } ua;
        ua.q[0] = *(const uint4*)(arow + kb * 32 + hi * 8);
        ua.q[1] = *(const uint4*)(arow + kb * 32 + 16 + hi * 8);
        afr[kb] = ua.v;
    }
    __syncthreads();   // A reads done; shared pool may now hold B tiles

    // per-wave row constants (C layout: value r of v8f is row r + 8*hi)
    int   spk8[8];
    float dg8[8];
#pragma unroll
    for (int r = 0; r < 8; ++r) {
        const int gr = row0 + rt * 16 + hi * 8 + r;
        spk8[r] = gr / G_UTT;
        dg8[r]  = diag[gr];
    }
    float m8[8], s8[8];
    int   a8[8];
#pragma unroll
    for (int r = 0; r < 8; ++r) { m8[r] = -1e30f; s8[r] = 0.f; a8[r] = 0; }

    _Float16* ldsBcg = (_Float16*)smem4 + (size_t)cg * 16 * D_FEAT;
    const int lt = tid & 63;   // lane within column group (2 waves)

    for (int t = 0; t < TILES_PER_CG; ++t) {
        const int ct   = cg * TILES_PER_CG + t;
        const int col0 = ct * 16;

        // cooperative B-tile stage: 16 centroid rows x 512 f16 = 16KB = 1024 x b128
        const uint4* src = (const uint4*)(Bmat + (size_t)col0 * D_FEAT);
        uint4* dst = (uint4*)ldsBcg;
        for (int i = 0; i < 16; ++i) {
            const int idx = lt + i * 64;
            dst[idx] = src[idx];
        }
        __syncthreads();   // B visible to both row-waves of this cg

        // K-loop: B fragment double-buffered in registers so the DS loads for
        // step kb+1 are in flight while the WMMA for step kb executes.
        const _Float16* bbase = ldsBcg + (size_t)ln * D_FEAT + hi * 8;
        union UB { uint4 q[2]; v16h v; };
        UB buf[2];
        {
            const uint4* p0 = (const uint4*)bbase;
            buf[0].q[0] = p0[0];
            buf[0].q[1] = p0[2];
        }
        v8f c = {};
#pragma unroll
        for (int kb = 0; kb < KSTEPS; ++kb) {
            if (kb + 1 < KSTEPS) {
                const uint4* pn = (const uint4*)(bbase + (kb + 1) * 32);
                buf[(kb + 1) & 1].q[0] = pn[0];
                buf[(kb + 1) & 1].q[1] = pn[2];
            }
            c = __builtin_amdgcn_wmma_f32_16x16x32_f16(
                    false, afr[kb], false, buf[kb & 1].v, (short)0, c, false, false);
        }
        __syncthreads();   // all reads of ldsB done before next overwrite

        // epilogue: diag substitution, clip, affine, fixed-shift exp-sum + argmax
        const int colspk = col0 + ln;
#pragma unroll
        for (int r = 0; r < 8; ++r) {
            float v = c[r];
            if (colspk == spk8[r]) v = dg8[r];      // leave-one-out cosine
            v = fmaf(fmaxf(v, 1e-6f), w, bb);       // clip + affine
            s8[r] += __expf(v - shift);             // bounded: v <= shift
            if (v > m8[r]) { m8[r] = v; a8[r] = colspk; }
        }
    }

    // merge across the 16 lanes that share the same 8 rows (xor stays in half)
#pragma unroll
    for (int mask = 1; mask < 16; mask <<= 1) {
#pragma unroll
        for (int r = 0; r < 8; ++r) {
            s8[r] += __shfl_xor(s8[r], mask);
            const float mo = __shfl_xor(m8[r], mask);
            const int   ao = __shfl_xor(a8[r], mask);
            if (mo > m8[r]) { m8[r] = mo; a8[r] = ao; }
        }
    }
    if (ln == 0) {
#pragma unroll
        for (int r = 0; r < 8; ++r) {
            const int rl = rt * 16 + hi * 8 + r;
            smax[rl][cg] = m8[r];
            ssum[rl][cg] = s8[r];
            sarg[rl][cg] = a8[r];
        }
    }
    __syncthreads();

    // merge the 4 column-group partials per row; compute loss + precision
    if (tid < ROWS_PER_BLK) {
        float M = smax[tid][0], S = ssum[tid][0];
        int   A = sarg[tid][0];
        for (int gq = 1; gq < CGROUPS; ++gq) {
            S += ssum[tid][gq];
            const float mo = smax[tid][gq];
            const int   ao = sarg[tid][gq];
            if (mo > M) { M = mo; A = ao; }
        }
        const int gr  = row0 + tid;
        const int spk = gr / G_UTT;
        const float tgt = fmaf(fmaxf(diag[gr], 1e-6f), w, bb);
        lred[tid] = (shift + __logf(S)) - tgt;   // -log_softmax at label
        cred[tid] = (A == spk) ? 1.0f : 0.0f;
    }
    __syncthreads();
    if (tid == 0) {
        float L = 0.f, C = 0.f;
        for (int i = 0; i < ROWS_PER_BLK; ++i) { L += lred[i]; C += cred[i]; }
        atomicAdd(&accum[0], L);
        atomicAdd(&accum[1], C);
    }
}

// ---------------------------------------------------------------------------
// Kernel 3: finalize the two scalars
// ---------------------------------------------------------------------------
__global__ void ge2e_finalize_kernel(const float* __restrict__ accum,
                                     float* __restrict__ out) {
    if (threadIdx.x == 0 && blockIdx.x == 0) {
        out[0] = accum[0] * (1.0f / (float)NG);          // nloss
        out[1] = accum[1] * (100.0f / (float)NG);        // prec1
    }
}

// ---------------------------------------------------------------------------
// Workspace layout (all 256B-aligned):
//   [0)        2 floats: loss / correct accumulators
//   [256)      inv_x : NG floats (81920 B)
//   [82176)    diag  : NG floats (81920 B)
//   [164096)   Bmat  : N*D f16   (2 MB)
// ---------------------------------------------------------------------------
extern "C" void kernel_launch(void* const* d_in, const int* in_sizes, int n_in,
                              void* d_out, int out_size, void* d_ws, size_t ws_size,
                              hipStream_t stream) {
    const float* x  = (const float*)d_in[0];
    const float* wg = (const float*)d_in[1];
    const float* bb = (const float*)d_in[2];
    float* out = (float*)d_out;

    char* ws = (char*)d_ws;
    float*    accum = (float*)ws;
    float*    inv_x = (float*)(ws + 256);
    float*    diag  = (float*)(ws + 256 + (size_t)NG * 4);
    _Float16* Bmat  = (_Float16*)(ws + 256 + (size_t)2 * NG * 4);

    ge2e_init_kernel<<<1, 32, 0, stream>>>(accum);
    ge2e_prep_kernel<<<N_SPK, 256, 0, stream>>>(x, inv_x, diag, Bmat);
    ge2e_gemm_loss_kernel<<<NG / ROWS_PER_BLK, 256, 0, stream>>>(
        x, inv_x, diag, Bmat, wg, bb, accum);
    ge2e_finalize_kernel<<<1, 32, 0, stream>>>(accum, out);
}
